// GakeModel_54211077210506
// MI455X (gfx1250) — matrix operand order repaired and checked by reading, not compiled, not run on hardware
//
#include <hip/hip_runtime.h>

// GAKE loss on MI455X (gfx1250, wave32, WMMA).
// score_unnorm[k] = C_k . S, S = sum_j m_j C_j, computed as a WMMA matvec:
// A = 16-row tiles of the zero-padded bf16 context matrix, B = S broadcast
// across all 16 columns -> v_wmma_f32_16x16x32_bf16 yields final scores in
// the accumulator with no cross-lane reductions at all.

typedef __attribute__((ext_vector_type(16))) __bf16 v16bf;
typedef __attribute__((ext_vector_type(8)))  __bf16 v8bf;
typedef __attribute__((ext_vector_type(4)))  __bf16 v4bf;
typedef __attribute__((ext_vector_type(8)))  float  v8f;

#define TPB    256
#define DIM    128
#define MAXPAD 80   // max padded context rows: 1+64 -> 80

__launch_bounds__(TPB)
__global__ void gake_loss_kernel(
    const float* __restrict__ ent_emb,
    const float* __restrict__ rel_emb,
    const int* __restrict__ node_ids,
    const unsigned char* __restrict__ node_is_rel,
    const int* __restrict__ nbr_ids,
    const unsigned char* __restrict__ nbr_is_rel,
    const unsigned char* __restrict__ nbr_mask,
    const int* __restrict__ path_ids,
    const unsigned char* __restrict__ path_is_rel,
    const unsigned char* __restrict__ path_mask,
    const int* __restrict__ edge_ids,
    const unsigned char* __restrict__ edge_is_rel,
    const unsigned char* __restrict__ edge_mask,
    float* __restrict__ out,
    int KN, int KP, int KE, int R)
{
    __shared__ __attribute__((aligned(32))) __bf16 Cbh[MAXPAD * DIM]; // 20 KB
    __shared__ __attribute__((aligned(32))) __bf16 Sb[DIM];  // bf16 row-sum
    __shared__ float Sf[DIM];       // f32 row-sum accumulator
    __shared__ float mSh[MAXPAD];   // per-row validity (1.0 / 0.0)
    __shared__ float sSh[MAXPAD];   // unnormalized scores
    __shared__ float norm2Sh;       // ||C_masked||_F^2 (f32-accurate)
    __shared__ float lossSh;

    const int b    = blockIdx.x;
    const int tid  = threadIdx.x;
    const int lane = tid & 31;
    const int wave = tid >> 5;
    const int hi   = lane >> 4;     // half-wave selector for WMMA layouts
    const int lr   = lane & 15;

    if (tid == 0) lossSh = 0.f;

    const int*           idsA[3] = {nbr_ids,    path_ids,    edge_ids};
    const unsigned char* relA[3] = {nbr_is_rel, path_is_rel, edge_is_rel};
    const unsigned char* mskA[3] = {nbr_mask,   path_mask,   edge_mask};
    const int            KA[3]   = {KN, KP, KE};
    const float          wA[3]   = {1.0f, 0.1f, 0.1f};

    for (int phase = 0; phase < 3; ++phase) {
        const int K     = KA[phase];
        const int Kc    = K + 1;                 // +1 for the si row
        const int KcPad = ((Kc + 15) >> 4) << 4; // 80, 80, 32
        const int TI    = KcPad >> 4;            // number of 16-row tiles
        const int* cids = idsA[phase];
        const unsigned char* crel = relA[phase];
        const unsigned char* cmsk = mskA[phase];

        if (tid < DIM) Sf[tid] = 0.f;
        if (tid == 0) norm2Sh = 0.f;
        __syncthreads();

        // ---- Phase A: gather rows (mask-skipped), convert to bf16 in LDS.
        //      Each wave owns whole rows (lane = float4 column), so the row
        //      sum S and the masked Frobenius norm^2 accumulate for free in
        //      registers (masked/padded rows are zero and self-cancel). ----
        {
            float4 vacc = make_float4(0.f, 0.f, 0.f, 0.f);
            float  dacc = 0.f;
            for (int r0 = 0; r0 < KcPad; r0 += TPB / 32) {
                const int row = r0 + wave;
                float4 v = make_float4(0.f, 0.f, 0.f, 0.f);
                float mv = 0.f;
                if (row < Kc) {
                    int id = 0, rel = 0, mk = 0;
                    if (lane == 0) {
                        if (row == 0) {
                            id = node_ids[b]; rel = node_is_rel[b]; mk = 1;
                        } else {
                            const size_t j = (size_t)b * K + (row - 1);
                            id  = cids[j];
                            rel = crel[j];
                            mk  = cmsk[j] ? 1 : 0;
                        }
                    }
                    id  = __shfl(id, 0);
                    rel = __shfl(rel, 0);
                    mk  = __shfl(mk, 0);
                    mv  = (float)mk;
                    if (mk) {   // masked-out rows contribute nothing: skip load
                        const float* src = rel
                            ? (rel_emb + (size_t)((unsigned)id % (unsigned)R) * DIM)
                            : (ent_emb + (size_t)id * DIM);
                        v = ((const float4*)src)[lane];
                    }
                }
                v4bf o;
                o[0] = (__bf16)v.x; o[1] = (__bf16)v.y;
                o[2] = (__bf16)v.z; o[3] = (__bf16)v.w;
                *(v4bf*)(&Cbh[row * DIM + lane * 4]) = o;
                if (lane == 0) mSh[row] = mv;
                vacc.x += v.x; vacc.y += v.y; vacc.z += v.z; vacc.w += v.w;
                dacc += v.x * v.x + v.y * v.y + v.z * v.z + v.w * v.w;
            }
            // column sums -> Sf (8-way conflicts at most)
            atomicAdd(&Sf[lane * 4 + 0], vacc.x);
            atomicAdd(&Sf[lane * 4 + 1], vacc.y);
            atomicAdd(&Sf[lane * 4 + 2], vacc.z);
            atomicAdd(&Sf[lane * 4 + 3], vacc.w);
            // one reduction for the whole norm^2 contribution of this wave
            dacc += __shfl_xor(dacc, 16);
            dacc += __shfl_xor(dacc, 8);
            dacc += __shfl_xor(dacc, 4);
            dacc += __shfl_xor(dacc, 2);
            dacc += __shfl_xor(dacc, 1);
            if (lane == 0) atomicAdd(&norm2Sh, dacc);
        }
        __syncthreads();

        if (tid < DIM) Sb[tid] = (__bf16)Sf[tid];
        __syncthreads();

        // ---- Phase B: scores via WMMA matvec. A = rows ib..ib+15 of C,
        //      B = S replicated in all 16 columns. acc[r] then directly holds
        //      score_unnorm[ib + r + 8*hi] (identical across N). ----
        if (wave < TI) {
            const int ib = wave << 4;
            const __bf16* arow = &Cbh[(ib + lr) * DIM];
            v8f acc = {0.f, 0.f, 0.f, 0.f, 0.f, 0.f, 0.f, 0.f};
            #pragma unroll
            for (int kb = 0; kb < DIM; kb += 32) {
                // A 16x32 bf16: lane holds K = kb+16*hi+{0..15}
                // for row M = ib + lr.
                const v16bf av = *(const v16bf*)(&arow[kb + 16 * hi]);
                // B 32x16 bf16: lane holds K = kb+16*hi+{0..15}; every column
                // N gets the same S (wave-uniform LDS read -> broadcast).
                const v16bf bv = *(const v16bf*)(&Sb[kb + 16 * hi]);
                acc = __builtin_amdgcn_wmma_f32_16x16x32_bf16(
                    false, av, false, bv, (short)0, acc, false, false);
            }
            if (lr == 0) {
                #pragma unroll
                for (int r = 0; r < 8; ++r)
                    sSh[ib + r + 8 * hi] = acc[r];
            }
        }
        __syncthreads();

        // ---- Phase C: masked log-softmax + reduction (wave 0 only) ----
        if (wave == 0) {
            const float inv = rsqrtf(fmaxf(norm2Sh, 1e-30f));
            float smax = -3.402823466e+38f;
            for (int k = lane; k < MAXPAD; k += 32) {
                const bool valid = (k < Kc) && (mSh[k] > 0.5f);
                if (valid) smax = fmaxf(smax, sSh[k] * inv);
            }
            smax = fmaxf(smax, __shfl_xor(smax, 16));
            smax = fmaxf(smax, __shfl_xor(smax, 8));
            smax = fmaxf(smax, __shfl_xor(smax, 4));
            smax = fmaxf(smax, __shfl_xor(smax, 2));
            smax = fmaxf(smax, __shfl_xor(smax, 1));

            float se = 0.f, ss = 0.f, nv = 0.f;
            for (int k = lane; k < MAXPAD; k += 32) {
                const bool valid = (k < Kc) && (mSh[k] > 0.5f);
                if (valid) {
                    const float s = sSh[k] * inv;
                    se += __expf(s - smax);
                    ss += s;
                    nv += 1.f;
                }
            }
            #pragma unroll
            for (int m = 16; m >= 1; m >>= 1) {
                se += __shfl_xor(se, m);
                ss += __shfl_xor(ss, m);
                nv += __shfl_xor(nv, m);
            }
            const float logZ = __logf(se);
            float res;
            if (phase < 2) res = ss - nv * (smax + logZ);     // accumulate=True
            else           res = sSh[0] * inv - smax - logZ;  // logp[:,0]
            if (lane == 0) lossSh += wA[phase] * res;
        }
        __syncthreads();
    }

    if (tid == 0) out[b] = -lossSh;
}

extern "C" void kernel_launch(void* const* d_in, const int* in_sizes, int n_in,
                              void* d_out, int out_size, void* d_ws, size_t ws_size,
                              hipStream_t stream) {
    const float*         ent_emb     = (const float*)d_in[0];
    const float*         rel_emb     = (const float*)d_in[1];
    const int*           node_ids    = (const int*)d_in[2];
    const unsigned char* node_is_rel = (const unsigned char*)d_in[3];
    const int*           nbr_ids     = (const int*)d_in[4];
    const unsigned char* nbr_is_rel  = (const unsigned char*)d_in[5];
    const unsigned char* nbr_mask    = (const unsigned char*)d_in[6];
    const int*           path_ids    = (const int*)d_in[7];
    const unsigned char* path_is_rel = (const unsigned char*)d_in[8];
    const unsigned char* path_mask   = (const unsigned char*)d_in[9];
    const int*           edge_ids    = (const int*)d_in[10];
    const unsigned char* edge_is_rel = (const unsigned char*)d_in[11];
    const unsigned char* edge_mask   = (const unsigned char*)d_in[12];
    float*               out         = (float*)d_out;

    const int Bn = in_sizes[2];
    const int KN = in_sizes[4]  / Bn;
    const int KP = in_sizes[7]  / Bn;
    const int KE = in_sizes[10] / Bn;
    const int R  = in_sizes[1] / DIM;

    gake_loss_kernel<<<Bn, TPB, 0, stream>>>(
        ent_emb, rel_emb, node_ids, node_is_rel,
        nbr_ids, nbr_is_rel, nbr_mask,
        path_ids, path_is_rel, path_mask,
        edge_ids, edge_is_rel, edge_mask,
        out, KN, KP, KE, R);
}
